// KANLayer_30777735643159
// MI455X (gfx1250) — matrix operand order repaired and checked by reading, not compiled, not run on hardware
//
#include <hip/hip_runtime.h>
#include <hip/hip_bf16.h>

// ---------------------------------------------------------------------------
// KAN layer, fused as ONE bf16 WMMA GEMM on gfx1250 (wave32, 16x16x32 bf16).
//
//   A_aug (4096 x 1024) = [ X | tanh(X) | tanh^2 | tanh^3 ]      (bf16)
//   W_aug (1024 x  256) = [ Wb ; g*S1 ; g*S2 ; g*S3 ]            (bf16)
//   cvec  (256)         = bias + sum_i gate[i,j]*spline[i,j,0]   (f32)
//   out = selu(A_aug @ W_aug + cvec)                             (f32)
//
// A_aug and W_aug are written by the prep kernels directly in the per-lane
// WMMA fragment layouts (ISA 7.12.2), so the GEMM does only coalesced
// 32B/lane fragment loads + v_wmma_f32_16x16x32_bf16.
//
// No explicit prefetch: the working set is GL2-resident, loads are clause-
// issued a full WMMA-group ahead, and 8 waves/block hide the rest. Probing
// showed both __builtin_prefetch (SE/SYS scope only -> no WGP prefetch) and
// inline-asm WGP prefetch (asm barrier + un-foldable address math) degrade
// the schedule for zero data-movement benefit.
// ---------------------------------------------------------------------------

typedef __attribute__((ext_vector_type(16))) __bf16 v16bf;
typedef __attribute__((ext_vector_type(8)))  float  v8f;

#define BATCH   4096
#define DIM     256
#define UNITS   256
#define KAUG    1024          // 4 * DIM
#define KB      32            // KAUG / 32  (K-blocks of 32)

// --------------------------- prep: A fragments ------------------------------
// A-frag element index = ((mb*KB + kb)*32 + lane)*16 + e
// 16-bit A-matrix 16x32 layout: lanes 0-15: v0-3 hold K0-7, v4-7 hold K16-23;
// lanes 16-31: v0-3 hold K8-15, v4-7 hold K24-31.
__global__ __launch_bounds__(256) void kan_prep_A(
    const float* __restrict__ X, __bf16* __restrict__ Abuf)
{
    int tid  = blockIdx.x * 256 + threadIdx.x;
    int e    = tid & 15;
    int lane = (tid >> 4) & 31;
    int f    = tid >> 9;
    int kb   = f & (KB - 1);
    int mb   = f >> 5;
    int hi   = lane >> 4;
    int kin  = (e < 8) ? (hi * 8 + e) : (8 + hi * 8 + e);  // A packing
    int k    = kb * 32 + kin;
    int d    = k >> 8;       // which power: 0..3
    int i    = k & 255;      // input feature
    int m    = mb * 16 + (lane & 15);

    float x = X[m * DIM + i];
    float v;
    if (d == 0) {
        v = x;
    } else {
        float t = tanhf(x);
        v = t;
        if (d >= 2) v *= t;
        if (d == 3) v *= t;
    }
    Abuf[tid] = (__bf16)v;
}

// --------------------------- prep: B fragments ------------------------------
// B-frag element index = ((nb*KB + kb)*32 + lane)*16 + e
// 16-bit B-matrix KxN layout: lanes 0-15 hold K0-7, K16-23 (2/VGPR), lanes 16-31
// hold K8-15, K24-31; each lane holds one column n = lane%16.
__global__ __launch_bounds__(256) void kan_prep_W(
    const float* __restrict__ Wb, const float* __restrict__ spline,
    const float* __restrict__ gate, __bf16* __restrict__ Wbuf)
{
    int tid  = blockIdx.x * 256 + threadIdx.x;
    int e    = tid & 15;
    int lane = (tid >> 4) & 31;
    int f    = tid >> 9;
    int kb   = f & (KB - 1);
    int nb   = f >> 5;
    int hi   = lane >> 4;
    int kin  = (e < 8) ? (hi * 8 + e) : (8 + hi * 8 + e);  // B packing (same as A)
    int k    = kb * 32 + kin;
    int d    = k >> 8;
    int i    = k & 255;
    int j    = nb * 16 + (lane & 15);

    float v = (d == 0) ? Wb[i * UNITS + j]
                       : gate[i * UNITS + j] * spline[(i * UNITS + j) * 4 + d];
    Wbuf[tid] = (__bf16)v;
}

// ------------------- prep: column constant (bias + d=0 term) ----------------
__global__ __launch_bounds__(256) void kan_prep_cvec(
    const float* __restrict__ bias, const float* __restrict__ spline,
    const float* __restrict__ gate, float* __restrict__ cvec)
{
    int j = threadIdx.x;
    float s = bias[j];
    for (int i = 0; i < DIM; ++i)
        s += gate[i * UNITS + j] * spline[(i * UNITS + j) * 4 + 0];
    cvec[j] = s;
}

// ------------------------------ WMMA GEMM -----------------------------------
// Block = 8 waves, computes 128 rows x 64 cols of C.
// Wave w: m-tiles { blockIdx.y*8 + (w&3)*2 + {0,1} },
//         n-tiles { blockIdx.x*4 + (w>>2)*2 + {0,1} }.
__device__ __forceinline__ float selu_f(float x) {
    // scale = 1.0507009873554805, scale*alpha = 1.7580993408473766
    return x > 0.0f ? 1.0507009873554805f * x
                    : 1.7580993408473766f * (__expf(x) - 1.0f);
}

__global__ __launch_bounds__(256) void kan_wmma_gemm(
    const __bf16* __restrict__ Abuf, const __bf16* __restrict__ Wbuf,
    const float* __restrict__ cvec, float* __restrict__ out)
{
    const int lane = threadIdx.x & 31;
    const int wave = threadIdx.x >> 5;

    const int mb0 = blockIdx.y * 8 + (wave & 3) * 2;   // global m-tile base
    const int nb0 = blockIdx.x * 4 + (wave >> 2) * 2;  // global n-tile base

    const v16bf* __restrict__ Af = (const v16bf*)Abuf;
    const v16bf* __restrict__ Bf = (const v16bf*)Wbuf;

    // fragment indices in v16bf units; step per kb is 32 (one frag = 32 lanes)
    long a0 = ((long)(mb0 + 0) * KB) * 32 + lane;
    long a1 = ((long)(mb0 + 1) * KB) * 32 + lane;
    long b0 = ((long)(nb0 + 0) * KB) * 32 + lane;
    long b1 = ((long)(nb0 + 1) * KB) * 32 + lane;

    v8f acc00 = {}, acc01 = {}, acc10 = {}, acc11 = {};

#pragma unroll 4
    for (int kb = 0; kb < KB; ++kb) {
        v16bf va0 = Af[a0 + kb * 32];
        v16bf va1 = Af[a1 + kb * 32];
        v16bf vb0 = Bf[b0 + kb * 32];
        v16bf vb1 = Bf[b1 + kb * 32];

        acc00 = __builtin_amdgcn_wmma_f32_16x16x32_bf16(
            false, va0, false, vb0, (short)0, acc00, false, false);
        acc01 = __builtin_amdgcn_wmma_f32_16x16x32_bf16(
            false, va0, false, vb1, (short)0, acc01, false, false);
        acc10 = __builtin_amdgcn_wmma_f32_16x16x32_bf16(
            false, va1, false, vb0, (short)0, acc10, false, false);
        acc11 = __builtin_amdgcn_wmma_f32_16x16x32_bf16(
            false, va1, false, vb1, (short)0, acc11, false, false);
    }

    // Epilogue: C/D 16x16 f32 layout — VGPR r: lanes 0-15 -> (M=r, N=lane),
    // lanes 16-31 -> (M=8+r, N=lane-16).
    const int colhalf = lane & 15;
    const int rowoff  = (lane >> 4) * 8;

    {
        int col = (nb0 + 0) * 16 + colhalf;
        float c = cvec[col];
        int rb0 = (mb0 + 0) * 16 + rowoff;
        int rb1 = (mb0 + 1) * 16 + rowoff;
#pragma unroll
        for (int r = 0; r < 8; ++r) {
            out[(long)(rb0 + r) * UNITS + col] = selu_f(acc00[r] + c);
            out[(long)(rb1 + r) * UNITS + col] = selu_f(acc10[r] + c);
        }
    }
    {
        int col = (nb0 + 1) * 16 + colhalf;
        float c = cvec[col];
        int rb0 = (mb0 + 0) * 16 + rowoff;
        int rb1 = (mb0 + 1) * 16 + rowoff;
#pragma unroll
        for (int r = 0; r < 8; ++r) {
            out[(long)(rb0 + r) * UNITS + col] = selu_f(acc01[r] + c);
            out[(long)(rb1 + r) * UNITS + col] = selu_f(acc11[r] + c);
        }
    }
}

// ------------------------------- launcher -----------------------------------
extern "C" void kernel_launch(void* const* d_in, const int* in_sizes, int n_in,
                              void* d_out, int out_size, void* d_ws, size_t ws_size,
                              hipStream_t stream)
{
    (void)in_sizes; (void)n_in; (void)out_size; (void)ws_size;

    const float* X      = (const float*)d_in[0];  // (4096, 256)
    const float* Wb     = (const float*)d_in[1];  // (256, 256)
    const float* bias   = (const float*)d_in[2];  // (256,)
    const float* spline = (const float*)d_in[3];  // (256, 256, 4)
    const float* gate   = (const float*)d_in[4];  // (256, 256)
    float* out          = (float*)d_out;          // (4096, 256)

    // workspace: A frags (8 MB bf16) | W frags (512 KB bf16) | cvec (1 KB f32)
    __bf16* Abuf = (__bf16*)d_ws;
    __bf16* Wbuf = Abuf + (size_t)BATCH * KAUG;
    float*  cvec = (float*)(Wbuf + (size_t)KAUG * UNITS);

    kan_prep_A<<<(BATCH * KAUG) / 256, 256, 0, stream>>>(X, Abuf);
    kan_prep_W<<<(KAUG * UNITS) / 256, 256, 0, stream>>>(Wb, spline, gate, Wbuf);
    kan_prep_cvec<<<1, 256, 0, stream>>>(bias, spline, gate, cvec);

    dim3 grid(UNITS / 64, BATCH / 128);   // (4, 32)
    kan_wmma_gemm<<<grid, 256, 0, stream>>>(Abuf, Wbuf, cvec, out);
}
